// GeometricTransportReadingBlock_44796508897647
// MI455X (gfx1250) — compile-verified, matrix operand order checked by reading, not run on hardware
//
#include <hip/hip_runtime.h>
#include <hip/hip_bf16.h>

#define NHD 8
#define HDD 64
#define DIM 512
#define BB  8
#define TT  1024
#define SS  1024
#define CC  512

typedef __attribute__((ext_vector_type(16))) __bf16 bf16x16;
typedef __attribute__((ext_vector_type(8)))  float  f32x8;

union BfVec {
  unsigned int   d[8];
  unsigned short u[16];
  bf16x16        v;
};

__device__ __forceinline__ unsigned short f2bf(float x) {
  unsigned int u = __float_as_uint(x);
  u += 0x7FFFu + ((u >> 16) & 1u);   // round-to-nearest-even
  return (unsigned short)(u >> 16);
}

// K-offset for dword j (pair of consecutive K elems) in the 16-bit A/B WMMA
// operand layout (CDNA5 ISA, 16-bit A 16x32 table).
__device__ __forceinline__ int koff(int j, int hf) {
  return ((j & 4) << 2) + hf * 8 + ((2 * j) & 7);
}

// ---------------------------------------------------------------------------
// Blocked bf16 GEMM: C(MxN,f32) = A(MxK,bf16) * BT(NxK,bf16)^T + bias, opt SiLU.
// One wave computes a 32x64 macro-tile (2x4 WMMA tiles): A/B operands loaded
// once per K-step feed 8 WMMAs -> ~2.7x less L2 traffic than 16x16/wave.
// ---------------------------------------------------------------------------
__global__ void k_gemm_bf16(const unsigned short* __restrict__ A,
                            const unsigned short* __restrict__ BT,
                            const float* __restrict__ bias,
                            float* __restrict__ C,
                            int M, int N, int K, int act) {
  int wave = blockIdx.x * (blockDim.x >> 5) + (threadIdx.x >> 5);
  int ntn = N >> 6;                       // 64-wide super-cols
  int total = (M >> 5) * ntn;             // 32-tall super-rows
  if (wave >= total) return;
  int tm = wave / ntn, tn = wave % ntn;
  int lane = threadIdx.x & 31;
  int hf = lane >> 4, idx = lane & 15;

  const unsigned short* arow = A  + (size_t)(tm * 32 + idx) * K;
  const unsigned short* brow = BT + (size_t)(tn * 64 + idx) * K;
  f32x8 acc[2][4] = {};
  for (int k0 = 0; k0 < K; k0 += 32) {
    BfVec a[2], bv[4];
#pragma unroll
    for (int j = 0; j < 8; ++j) {
      int o = k0 + koff(j, hf);
      a[0].d[j]  = *(const unsigned int*)(arow + o);
      a[1].d[j]  = *(const unsigned int*)(arow + (size_t)16 * K + o);
      bv[0].d[j] = *(const unsigned int*)(brow + o);
      bv[1].d[j] = *(const unsigned int*)(brow + (size_t)16 * K + o);
      bv[2].d[j] = *(const unsigned int*)(brow + (size_t)32 * K + o);
      bv[3].d[j] = *(const unsigned int*)(brow + (size_t)48 * K + o);
    }
    if (k0 + 32 < K) {                    // near-cache prefetch (global_prefetch_b8)
      __builtin_prefetch(arow + k0 + 32, 0, 3);
      __builtin_prefetch(arow + (size_t)16 * K + k0 + 32, 0, 3);
      __builtin_prefetch(brow + k0 + 32, 0, 3);
      __builtin_prefetch(brow + (size_t)32 * K + k0 + 32, 0, 3);
    }
#pragma unroll
    for (int i = 0; i < 2; ++i)
#pragma unroll
      for (int j = 0; j < 4; ++j)
        acc[i][j] = __builtin_amdgcn_wmma_f32_16x16x32_bf16(
            false, a[i].v, false, bv[j].v, (short)0, acc[i][j], false, false);
  }
#pragma unroll
  for (int i = 0; i < 2; ++i)
#pragma unroll
    for (int j = 0; j < 4; ++j) {
      int col = tn * 64 + j * 16 + idx;
      float bvs = bias ? bias[col] : 0.f;
      float* crow = C + (size_t)(tm * 32 + i * 16 + 8 * hf) * N + col;
#pragma unroll
      for (int r = 0; r < 8; ++r) {
        float x = acc[i][j][r] + bvs;
        if (act == 1) x = x / (1.f + __expf(-x));   // SiLU
        crow[(size_t)r * N] = x;
      }
    }
}

// ---------------------------------------------------------------------------
// Elementwise / layout kernels
// ---------------------------------------------------------------------------
__global__ void k_f32_to_bf16(const float* __restrict__ s,
                              unsigned short* __restrict__ d, int n) {
  int i = blockIdx.x * blockDim.x + threadIdx.x;
  if (i < n) d[i] = f2bf(s[i]);
}

// dst(N,K) bf16 = transpose of src(K,N) f32
__global__ void k_wT(const float* __restrict__ w, unsigned short* __restrict__ d,
                     int K, int N) {
  int i = blockIdx.x * blockDim.x + threadIdx.x;
  if (i >= N * K) return;
  int n = i / K, k = i % K;
  d[i] = f2bf(w[(size_t)k * N + n]);
}

// pl_w1 (6,512) -> (512,32) bf16, zero padded in K
__global__ void k_w1T_pad(const float* __restrict__ w, unsigned short* __restrict__ d) {
  int i = blockIdx.x * blockDim.x + threadIdx.x;
  if (i >= DIM * 32) return;
  int n = i >> 5, k = i & 31;
  d[i] = (k < 6) ? f2bf(w[(size_t)k * DIM + n]) : (unsigned short)0;
}

// plucker (B*T,6) -> (B*T,32) bf16 zero padded
__global__ void k_plucker_pad(const float* __restrict__ p, unsigned short* __restrict__ d) {
  int i = blockIdx.x * blockDim.x + threadIdx.x;
  if (i >= BB * TT * 32) return;
  int r = i >> 5, k = i & 31;
  d[i] = (k < 6) ? f2bf(p[(size_t)r * 6 + k]) : (unsigned short)0;
}

// front (B,C,T) f32 -> rows (B*T, C) bf16
__global__ void k_front_rows(const float* __restrict__ f, unsigned short* __restrict__ d) {
  int i = blockIdx.x * blockDim.x + threadIdx.x;
  if (i >= BB * TT * CC) return;
  int c = i % CC; int bt = i / CC; int t = bt % TT; int b = bt / TT;
  d[i] = f2bf(f[((size_t)b * CC + c) * TT + t]);
}

// (B*T, NH*HD) f32 -> (B,NH,T,HD) bf16
__global__ void k_pack_qk(const float* __restrict__ s, unsigned short* __restrict__ d) {
  int i = blockIdx.x * blockDim.x + threadIdx.x;
  if (i >= BB * TT * DIM) return;
  int col = i % DIM; int bt = i / DIM; int b = bt / TT; int t = bt % TT;
  int h = col >> 6; int dd = col & 63;
  d[(((size_t)(b * NHD + h)) * TT + t) * HDD + dd] = f2bf(s[i]);
}

// (B*S, NH*HD) f32 -> (B,NH,HD,S) bf16   (v transposed for the read WMMA)
__global__ void k_pack_vT(const float* __restrict__ s, unsigned short* __restrict__ d) {
  int i = blockIdx.x * blockDim.x + threadIdx.x;
  if (i >= BB * SS * DIM) return;
  int col = i % DIM; int bs = i / DIM; int b = bs / SS; int sidx = bs % SS;
  int h = col >> 6; int dd = col & 63;
  d[(((size_t)(b * NHD + h)) * HDD + dd) * SS + sidx] = f2bf(s[i]);
}

// ---------------------------------------------------------------------------
// LayerNorm over rows of 512; writes bf16 (and optional f32)
// ---------------------------------------------------------------------------
__global__ void k_layernorm(const float* __restrict__ x, const float* __restrict__ g,
                            const float* __restrict__ be,
                            unsigned short* __restrict__ ob, float* __restrict__ of) {
  __shared__ float red[128];
  int row = blockIdx.x, tid = threadIdx.x;
  const float* xr = x + (size_t)row * DIM;
  float v[4]; float s = 0.f, s2 = 0.f;
#pragma unroll
  for (int j = 0; j < 4; ++j) { v[j] = xr[tid + 128 * j]; s += v[j]; s2 += v[j] * v[j]; }
  red[tid] = s; __syncthreads();
  for (int o = 64; o; o >>= 1) { if (tid < o) red[tid] += red[tid + o]; __syncthreads(); }
  float mean = red[0] * (1.f / DIM); __syncthreads();
  red[tid] = s2; __syncthreads();
  for (int o = 64; o; o >>= 1) { if (tid < o) red[tid] += red[tid + o]; __syncthreads(); }
  float var = red[0] * (1.f / DIM) - mean * mean;
  float rstd = rsqrtf(var + 1e-5f);
#pragma unroll
  for (int j = 0; j < 4; ++j) {
    int c = tid + 128 * j;
    float y = (v[j] - mean) * rstd * g[c] + be[c];
    ob[(size_t)row * DIM + c] = f2bf(y);
    if (of) of[(size_t)row * DIM + c] = y;
  }
}

// ---------------------------------------------------------------------------
// context = front_embed + pl ; sigma, dustbin logits ; zero confidence
// ---------------------------------------------------------------------------
__global__ void k_context(const float* __restrict__ fe, const float* __restrict__ pl,
                          const float* __restrict__ sigw, const float* __restrict__ sigb,
                          const float* __restrict__ dbw, const float* __restrict__ dbb,
                          float* __restrict__ sigma, float* __restrict__ db,
                          float* __restrict__ conf) {
  __shared__ float red[128];
  int row = blockIdx.x, tid = threadIdx.x;
  const float* a = fe + (size_t)row * DIM;
  const float* p = pl + (size_t)row * DIM;
  float ssum = 0.f, dsum[NHD] = {};
#pragma unroll
  for (int j = 0; j < 4; ++j) {
    int c = tid + 128 * j;
    float x = a[c] + p[c];
    ssum += x * sigw[c];
#pragma unroll
    for (int h = 0; h < NHD; ++h) dsum[h] += x * dbw[(size_t)c * NHD + h];
  }
  red[tid] = ssum; __syncthreads();
  for (int o = 64; o; o >>= 1) { if (tid < o) red[tid] += red[tid + o]; __syncthreads(); }
  if (tid == 0) {
    float z = red[0] + sigb[0];
    sigma[row] = 0.03f + 0.32f / (1.f + __expf(-z));
    conf[row] = 0.f;
  }
  for (int h = 0; h < NHD; ++h) {
    __syncthreads();
    red[tid] = dsum[h]; __syncthreads();
    for (int o = 64; o; o >>= 1) { if (tid < o) red[tid] += red[tid + o]; __syncthreads(); }
    if (tid == 0) db[(size_t)row * NHD + h] = red[0] + dbb[h];
  }
}

// ---------------------------------------------------------------------------
// Fused flash attention with geo bias, query-validity mask and dustbin.
// One wave per (b, h, 32-query super-tile): two query tiles share every
// K-tile / V^T-tile operand load (halves L2 traffic, doubles WMMA per byte).
// scores^T (keys x queries) fp32 C layout maps IN-LANE to the bf16 B-operand
// layout of probs^T for the out += V^T * probs^T WMMA (no cross-lane moves).
// 4 waves/block so a WGP's SIMDs share K/V lines in WGP$.
// ---------------------------------------------------------------------------
__global__ void __launch_bounds__(128)
k_attn(const unsigned short* __restrict__ Q,   // (B,NH,T,HD)
       const unsigned short* __restrict__ Km,  // (B,NH,S,HD)
       const unsigned short* __restrict__ VT,  // (B,NH,HD,S)
       const float* __restrict__ sigma,        // (B,T)
       const float* __restrict__ dbv,          // (B,T,NH)
       const float* __restrict__ bev_xy,       // (B,T,2)
       const float* __restrict__ sat_xy,       // (B,S,2)
       const unsigned char* __restrict__ valid,// (B,T)
       float* __restrict__ read,               // (B,T,NH*HD) f32
       float* __restrict__ conf) {             // (B,T) atomics
  int wv = blockIdx.x * (blockDim.x >> 5) + (threadIdx.x >> 5);
  if (wv >= BB * NHD * (TT / 32)) return;
  int ttile = wv & 31;                    // T/32 super-tiles
  int h = (wv >> 5) & 7;
  int b = wv >> 8;
  int lane = threadIdx.x & 31;
  int hf = lane >> 4, idx = lane & 15;
  size_t bh = (size_t)(b * NHD + h);

  int t[2]; float qx[2], qy[2], gsc[2]; bool vq[2];
  BfVec qb[2][2];
#pragma unroll
  for (int qt = 0; qt < 2; ++qt) {
    t[qt] = ttile * 32 + qt * 16 + idx;
    const unsigned short* qrow = Q + (bh * TT + t[qt]) * HDD;
#pragma unroll
    for (int j = 0; j < 8; ++j) {
      int o = koff(j, hf);
      qb[qt][0].d[j] = *(const unsigned int*)(qrow + o);
      qb[qt][1].d[j] = *(const unsigned int*)(qrow + 32 + o);
    }
    qx[qt] = bev_xy[((size_t)b * TT + t[qt]) * 2];
    qy[qt] = bev_xy[((size_t)b * TT + t[qt]) * 2 + 1];
    float sg = sigma[b * TT + t[qt]];
    gsc[qt] = -8.0f / fmaxf(sg * sg, 1e-6f);  // -LAMBDA_GEO / max(sigma^2,1e-6)
    vq[qt] = valid[b * TT + t[qt]] != 0;
  }
  float m[2] = {-1e30f, -1e30f}, l[2] = {0.f, 0.f};
  f32x8 acc[2][4] = {};
  const unsigned short* kbase = Km + bh * (size_t)SS * HDD;
  const unsigned short* vbase = VT + bh * (size_t)HDD * SS;
  const float* sxy = sat_xy + (size_t)b * SS * 2;

  for (int kb = 0; kb < SS; kb += 32) {
    // shared key A-operands (m = key, K = d): two key tiles x two d-chunks
    const unsigned short* k0r = kbase + (size_t)(kb + idx) * HDD;
    const unsigned short* k1r = k0r + 16 * HDD;
    BfVec a00, a01, a10, a11;
#pragma unroll
    for (int j = 0; j < 8; ++j) {
      int o = koff(j, hf);
      a00.d[j] = *(const unsigned int*)(k0r + o);
      a01.d[j] = *(const unsigned int*)(k0r + 32 + o);
      a10.d[j] = *(const unsigned int*)(k1r + o);
      a11.d[j] = *(const unsigned int*)(k1r + 32 + o);
    }
    BfVec pb[2];
#pragma unroll
    for (int qt = 0; qt < 2; ++qt) {
      f32x8 s0 = {}, s1 = {};
      s0 = __builtin_amdgcn_wmma_f32_16x16x32_bf16(false, a00.v, false, qb[qt][0].v, (short)0, s0, false, false);
      s0 = __builtin_amdgcn_wmma_f32_16x16x32_bf16(false, a01.v, false, qb[qt][1].v, (short)0, s0, false, false);
      s1 = __builtin_amdgcn_wmma_f32_16x16x32_bf16(false, a10.v, false, qb[qt][0].v, (short)0, s1, false, false);
      s1 = __builtin_amdgcn_wmma_f32_16x16x32_bf16(false, a11.v, false, qb[qt][1].v, (short)0, s1, false, false);

      // logits = 0.0125*qk + geo ; invalid query -> -10000 (dustbin untouched)
      float sc[16]; float mt = -1e30f;
#pragma unroll
      for (int r = 0; r < 8; ++r) {
        int key0 = kb + r + 8 * hf;
        float dx = qx[qt] - sxy[key0 * 2], dy = qy[qt] - sxy[key0 * 2 + 1];
        float v0 = vq[qt] ? (s0[r] * 0.0125f + gsc[qt] * (dx * dx + dy * dy)) : -10000.f;
        int key1 = key0 + 16;
        float dx1 = qx[qt] - sxy[key1 * 2], dy1 = qy[qt] - sxy[key1 * 2 + 1];
        float v1 = vq[qt] ? (s1[r] * 0.0125f + gsc[qt] * (dx1 * dx1 + dy1 * dy1)) : -10000.f;
        sc[r] = v0; sc[r + 8] = v1;
        mt = fmaxf(mt, fmaxf(v0, v1));
      }
      mt = fmaxf(mt, __shfl_xor(mt, 16, 32));     // merge lane-halves per query
      float mnew = fmaxf(m[qt], mt);
      float corr = __expf(m[qt] - mnew);
      float ps = 0.f;
#pragma unroll
      for (int p = 0; p < 16; ++p) {              // identity map: C layout -> B layout
        float e = __expf(sc[p] - mnew);
        ps += e;
        pb[qt].u[p] = f2bf(e);
      }
      ps += __shfl_xor(ps, 16, 32);
      l[qt] = l[qt] * corr + ps;
      m[qt] = mnew;
#pragma unroll
      for (int dc = 0; dc < 4; ++dc)
#pragma unroll
        for (int r = 0; r < 8; ++r) acc[qt][dc][r] *= corr;
    }
    // out^T += V^T(d x 32keys) * probs^T(32keys x 16q), V^T shared by both qt
#pragma unroll
    for (int dc = 0; dc < 4; ++dc) {
      const unsigned short* vr = vbase + (size_t)(dc * 16 + idx) * SS + kb;
      BfVec av;
#pragma unroll
      for (int j = 0; j < 8; ++j) av.d[j] = *(const unsigned int*)(vr + koff(j, hf));
      acc[0][dc] = __builtin_amdgcn_wmma_f32_16x16x32_bf16(false, av.v, false, pb[0].v,
                                                           (short)0, acc[0][dc], false, false);
      acc[1][dc] = __builtin_amdgcn_wmma_f32_16x16x32_bf16(false, av.v, false, pb[1].v,
                                                           (short)0, acc[1][dc], false, false);
    }
  }

  // dustbin as one extra logit; normalize; scatter read + confidence
#pragma unroll
  for (int qt = 0; qt < 2; ++qt) {
    float dl = dbv[((size_t)b * TT + t[qt]) * NHD + h];
    float mnew = fmaxf(m[qt], dl);
    float corr = __expf(m[qt] - mnew);
    float pd = __expf(dl - mnew);
    float lf = l[qt] * corr + pd;
    float rl = 1.f / lf;
    float scale = corr * rl;
    float* orow = read + ((size_t)(b * TT + t[qt])) * (NHD * HDD) + h * HDD;
#pragma unroll
    for (int dc = 0; dc < 4; ++dc)
#pragma unroll
      for (int r = 0; r < 8; ++r)
        orow[dc * 16 + 8 * hf + r] = acc[qt][dc][r] * scale;
    if (hf == 0)
      atomicAdd(&conf[b * TT + t[qt]], (1.f - pd * rl) * (1.f / NHD));
  }
}

// ---------------------------------------------------------------------------
// front_out = front + conf*read_feat ; also emit read_feat in (B,C,T)
// ---------------------------------------------------------------------------
__global__ void k_finalize(const float* __restrict__ front, const float* __restrict__ rf,
                           const float* __restrict__ conf,
                           float* __restrict__ out_front, float* __restrict__ out_rf) {
  int i = blockIdx.x * blockDim.x + threadIdx.x;
  if (i >= BB * CC * TT) return;
  int t = i % TT; int bc = i / TT; int c = bc % CC; int b = bc / CC;
  float r = rf[((size_t)b * TT + t) * CC + c];
  out_rf[i] = r;
  out_front[i] = front[i] + conf[b * TT + t] * r;
}

__global__ void k_loss(const float* __restrict__ conf, const unsigned char* __restrict__ valid,
                       float* __restrict__ out) {
  __shared__ float r1[256], r2[256];
  int tid = threadIdx.x;
  float s1 = 0.f, s2 = 0.f;
  for (int i = tid; i < BB * TT; i += 256) {
    float inv = valid[i] ? 0.f : 1.f;
    s1 += conf[i] * inv; s2 += inv;
  }
  r1[tid] = s1; r2[tid] = s2; __syncthreads();
  for (int o = 128; o; o >>= 1) {
    if (tid < o) { r1[tid] += r1[tid + o]; r2[tid] += r2[tid + o]; }
    __syncthreads();
  }
  if (tid == 0) out[0] = (r1[0] / fmaxf(r2[0], 1.f)) * 0.05f;
}

// ---------------------------------------------------------------------------
extern "C" void kernel_launch(void* const* d_in, const int* in_sizes, int n_in,
                              void* d_out, int out_size, void* d_ws, size_t ws_size,
                              hipStream_t stream) {
  (void)in_sizes; (void)n_in; (void)out_size; (void)ws_size;
  const float* front   = (const float*)d_in[0];
  const float* sat     = (const float*)d_in[1];
  const float* sat_xy  = (const float*)d_in[2];
  const float* bev_xy  = (const float*)d_in[3];
  const unsigned char* valid = (const unsigned char*)d_in[4];
  const float* pluck   = (const float*)d_in[5];
  const float* fa_w = (const float*)d_in[6],  *fa_b = (const float*)d_in[7];
  const float* fa_g = (const float*)d_in[8],  *fa_lb = (const float*)d_in[9];
  const float* sa_w = (const float*)d_in[10], *sa_b = (const float*)d_in[11];
  const float* sa_g = (const float*)d_in[12], *sa_lb = (const float*)d_in[13];
  const float* q_w  = (const float*)d_in[14], *q_b = (const float*)d_in[15];
  const float* k_w  = (const float*)d_in[16], *k_b = (const float*)d_in[17];
  const float* v_w  = (const float*)d_in[18], *v_b = (const float*)d_in[19];
  const float* pl_w1 = (const float*)d_in[20], *pl_b1 = (const float*)d_in[21];
  const float* pl_w2 = (const float*)d_in[22], *pl_b2 = (const float*)d_in[23];
  const float* sig_w = (const float*)d_in[24], *sig_b = (const float*)d_in[25];
  const float* db_w  = (const float*)d_in[26], *db_b  = (const float*)d_in[27];
  const float* out_w = (const float*)d_in[28], *out_b = (const float*)d_in[29];

  char* ws = (char*)d_ws;
  auto alloc = [&](size_t bytes) -> char* {
    char* p = ws; ws += (bytes + 255) & ~(size_t)255; return p;
  };
  const size_t R = (size_t)BB * TT;          // 8192 rows on both front & sat side
  unsigned short* faT  = (unsigned short*)alloc(DIM * DIM * 2);
  unsigned short* saT  = (unsigned short*)alloc(DIM * DIM * 2);
  unsigned short* qT   = (unsigned short*)alloc(DIM * DIM * 2);
  unsigned short* kT   = (unsigned short*)alloc(DIM * DIM * 2);
  unsigned short* vT_w = (unsigned short*)alloc(DIM * DIM * 2);
  unsigned short* pl2T = (unsigned short*)alloc(DIM * DIM * 2);
  unsigned short* outT = (unsigned short*)alloc(DIM * DIM * 2);
  unsigned short* pl1T = (unsigned short*)alloc(DIM * 32 * 2);
  unsigned short* frows = (unsigned short*)alloc(R * DIM * 2);
  unsigned short* satb  = (unsigned short*)alloc(R * DIM * 2);
  unsigned short* plpad = (unsigned short*)alloc(R * 32 * 2);
  float* scrA  = (float*)alloc(R * DIM * 4);   // reused: preLN / q,k,v / read
  float* scrB  = (float*)alloc(R * DIM * 4);   // reused: pl-hidden / read_feat
  float* fe_f  = (float*)alloc(R * DIM * 4);
  unsigned short* fe_b = (unsigned short*)alloc(R * DIM * 2);
  float* pl_f  = (float*)alloc(R * DIM * 4);
  unsigned short* h_b  = (unsigned short*)alloc(R * DIM * 2);  // also read_bf16
  unsigned short* se_b = (unsigned short*)alloc(R * DIM * 2);
  unsigned short* qp   = (unsigned short*)alloc(R * DIM * 2);
  unsigned short* kp   = (unsigned short*)alloc(R * DIM * 2);
  unsigned short* vtp  = (unsigned short*)alloc(R * DIM * 2);
  float* sigma = (float*)alloc(R * 4);
  float* dbv   = (float*)alloc(R * NHD * 4);

  float* out_front = (float*)d_out;
  float* out_rf    = out_front + (size_t)BB * CC * TT;
  float* out_conf  = out_rf + (size_t)BB * CC * TT;
  float* out_loss  = out_conf + R;

  auto gemm = [&](const unsigned short* A, const unsigned short* Bt, const float* bias,
                  float* C, int M, int N, int K, int act) {
    int tiles = (M / 32) * (N / 64);
    k_gemm_bf16<<<(tiles + 7) / 8, 256, 0, stream>>>(A, Bt, bias, C, M, N, K, act);
  };
  const int EW = 256;
  int nW = DIM * DIM;

  // weight prep (bf16, transposed to (N,K) for contiguous B-operand fetches)
  k_wT<<<(nW + EW - 1) / EW, EW, 0, stream>>>(fa_w, faT, DIM, DIM);
  k_wT<<<(nW + EW - 1) / EW, EW, 0, stream>>>(sa_w, saT, DIM, DIM);
  k_wT<<<(nW + EW - 1) / EW, EW, 0, stream>>>(q_w,  qT,  DIM, DIM);
  k_wT<<<(nW + EW - 1) / EW, EW, 0, stream>>>(k_w,  kT,  DIM, DIM);
  k_wT<<<(nW + EW - 1) / EW, EW, 0, stream>>>(v_w,  vT_w, DIM, DIM);
  k_wT<<<(nW + EW - 1) / EW, EW, 0, stream>>>(pl_w2, pl2T, DIM, DIM);
  k_wT<<<(nW + EW - 1) / EW, EW, 0, stream>>>(out_w, outT, DIM, DIM);
  k_w1T_pad<<<(DIM * 32 + EW - 1) / EW, EW, 0, stream>>>(pl_w1, pl1T);
  k_front_rows<<<(int)(R * DIM / EW), EW, 0, stream>>>(front, frows);
  k_f32_to_bf16<<<(int)(R * DIM / EW), EW, 0, stream>>>(sat, satb, (int)(R * DIM));
  k_plucker_pad<<<(int)(R * 32 / EW), EW, 0, stream>>>(pluck, plpad);

  // front embed -> LN (keep fp32 for context) -> bf16 for q
  gemm(frows, faT, fa_b, scrA, (int)R, DIM, DIM, 0);
  k_layernorm<<<(int)R, 128, 0, stream>>>(scrA, fa_g, fa_lb, fe_b, fe_f);
  // plucker MLP: silu(X@W1+b1)@W2+b2
  gemm(plpad, pl1T, pl_b1, scrB, (int)R, DIM, 32, 1);
  k_f32_to_bf16<<<(int)(R * DIM / EW), EW, 0, stream>>>(scrB, h_b, (int)(R * DIM));
  gemm(h_b, pl2T, pl_b2, pl_f, (int)R, DIM, DIM, 0);
  // sat embed -> LN (bf16 only)
  gemm(satb, saT, sa_b, scrA, (int)R, DIM, DIM, 0);
  k_layernorm<<<(int)R, 128, 0, stream>>>(scrA, sa_g, sa_lb, se_b, nullptr);
  // q / k / v projections + head-major packing
  gemm(fe_b, qT, q_b, scrA, (int)R, DIM, DIM, 0);
  k_pack_qk<<<(int)(R * DIM / EW), EW, 0, stream>>>(scrA, qp);
  gemm(se_b, kT, k_b, scrA, (int)R, DIM, DIM, 0);
  k_pack_qk<<<(int)(R * DIM / EW), EW, 0, stream>>>(scrA, kp);
  gemm(se_b, vT_w, v_b, scrA, (int)R, DIM, DIM, 0);
  k_pack_vT<<<(int)(R * DIM / EW), EW, 0, stream>>>(scrA, vtp);
  // context-derived sigma / dustbin logits, zero confidence accumulator
  k_context<<<(int)R, 128, 0, stream>>>(fe_f, pl_f, sig_w, sig_b, db_w, db_b,
                                        sigma, dbv, out_conf);
  // fused attention (scores never materialized); read -> scrA (b,t,h*64+d)
  k_attn<<<BB * NHD * (TT / 32) / 4, 128, 0, stream>>>(qp, kp, vtp, sigma, dbv,
                                                       bev_xy, sat_xy, valid,
                                                       scrA, out_conf);
  // out projection + final combine + loss
  k_f32_to_bf16<<<(int)(R * DIM / EW), EW, 0, stream>>>(scrA, h_b, (int)(R * DIM));
  gemm(h_b, outT, out_b, scrB, (int)R, CC, DIM, 0);
  k_finalize<<<(int)((size_t)BB * CC * TT / EW), EW, 0, stream>>>(front, scrB, out_conf,
                                                                  out_front, out_rf);
  k_loss<<<1, 256, 0, stream>>>(out_conf, valid, out_loss);
}
